// QuantInception_87016037417515
// MI455X (gfx1250) — matrix operand order repaired and checked by reading, not compiled, not run on hardware
//
#include <hip/hip_runtime.h>

typedef __attribute__((ext_vector_type(16))) _Float16 v16h;
typedef __attribute__((ext_vector_type(8)))  _Float16 v8h;
typedef __attribute__((ext_vector_type(8)))  float    v8f;
typedef __attribute__((ext_vector_type(4)))  unsigned int v4u;
typedef __attribute__((ext_vector_type(8)))  int      v8i;
typedef __attribute__((ext_vector_type(4)))  int      v4i;

static constexpr int kHW  = 28;
static constexpr int kPos = 28 * 28;   // 784
static constexpr int kImgs = 64;
static constexpr int kCoutTotal = 832;

// ---- Tensor Data Mover availability -----------------------------------------
#if defined(__has_builtin)
#  if __has_builtin(__builtin_amdgcn_tensor_load_to_lds) && \
      __has_builtin(__builtin_amdgcn_s_wait_tensorcnt)
#    define USE_TDM 1
#  endif
#endif
#ifndef USE_TDM
#  define USE_TDM 0
#endif
#if defined(__has_include)
#  if __has_include(<hip/amd_detail/amd_gfx1250_TDM.h>)
#    define TDM_SIX_ARGS 1
#  endif
#endif
#ifndef TDM_SIX_ARGS
#  define TDM_SIX_ARGS 0
#endif

// Issue a 1-D contiguous TDM copy (global -> LDS), nelem8 = count of 8-byte units.
// D# per CDNA5 ISA: group0 {count=1, lds_addr,
// global_addr, type=2}; group1 {data_size=3 (8B), tensor_dim0 = tile_dim0 =
// tensor_dim0_stride = nelem8, tensor_dim1 = 1}.
__device__ __forceinline__ void tdm_load_1d(const void* gsrc, void* lds, int nelem8) {
#if USE_TDM
  unsigned lds_addr = (unsigned)(__SIZE_TYPE__)lds;   // flat shared ptr: low bits = LDS byte addr
  unsigned long long ga = (unsigned long long)gsrc;
  v4u g0;
  g0[0] = 1u;                                         // count=1, user mode
  g0[1] = lds_addr;                                   // lds_addr [63:32]
  g0[2] = (unsigned)(ga & 0xFFFFFFFFu);               // global_addr [95:64]
  g0[3] = (unsigned)((ga >> 32) & 0x1FFFFFFu) | (2u << 30);  // addr hi | type=2
  v8i g1;
  g1[0] = 0x00030000;                                 // workgroup_mask=0, data_size=3 (8B)
  g1[1] = (int)(((unsigned)nelem8 & 0xFFFFu) << 16);  // tensor_dim0 lo -> [63:48]
  g1[2] = (int)((((unsigned)nelem8 >> 16) & 0xFFFFu) | (1u << 16)); // dim0 hi | tensor_dim1=1
  g1[3] = (int)(((unsigned)nelem8) << 16);            // tile_dim0 [127:112]
  g1[4] = 0;
  g1[5] = nelem8;                                     // tensor_dim0_stride lo
  g1[6] = 0;
  g1[7] = 0;
  v4i z4 = {};
#if TDM_SIX_ARGS
  v8i z8 = {};
  __builtin_amdgcn_tensor_load_to_lds(g0, g1, z4, z4, z8, 0);
#else
  __builtin_amdgcn_tensor_load_to_lds(g0, g1, z4, z4, 0);
#endif
#else
  (void)gsrc; (void)lds; (void)nelem8;
#endif
}

// ---------------------------------------------------------------------------
// Fragment loader for A (16Mx32K f16): lane<16 holds row M=lane,
// K in {k0..k0+15}, k0 = (lane&16)?16:0.  (contiguous k order, same as the B loads)
// ---------------------------------------------------------------------------
__device__ __forceinline__ v16h frag_a(const _Float16* base, int rstride, int lane) {
  int r  = lane & 15;
  int k0 = (lane & 16) ? 16 : 0;
  const _Float16* p = base + r * rstride;
  v8h lo = *(const v8h*)(p + k0);
  v8h hi = *(const v8h*)(p + k0 + 8);
  v16h f;
#pragma unroll
  for (int i = 0; i < 8; ++i) { f[i] = lo[i]; f[i + 8] = hi[i]; }
  return f;
}

// ---------------------------------------------------------------------------
// prep_input: fp32 NCHW x -> fp16 repacked [n][cb][pos][ci32] (xh) and the
// 3x3 maxpool of x in the same layout (xph). Thread id == repacked dst index.
// ---------------------------------------------------------------------------
__global__ void prep_input_kernel(const float* __restrict__ x,
                                  _Float16* __restrict__ xh,
                                  _Float16* __restrict__ xph,
                                  int total) {
  int t = blockIdx.x * blockDim.x + threadIdx.x;
  if (t >= total) return;
  int ci  = t & 31;
  int pos = (t >> 5) % kPos;
  int blk = (t >> 5) / kPos;      // n*16 + cb
  int n   = blk >> 4;
  int cb  = blk & 15;
  int c   = cb * 32 + ci;
  int w = pos % kHW, h = pos / kHW;
  size_t src = (size_t)(n * 512 + c) * kPos + pos;
  float cv = x[src];
  xh[t] = (_Float16)cv;
  float mx = cv;
#pragma unroll
  for (int dh = -1; dh <= 1; ++dh)
#pragma unroll
    for (int dw = -1; dw <= 1; ++dw) {
      int hh = h + dh, ww = w + dw;
      if (hh >= 0 && hh < kHW && ww >= 0 && ww < kHW)
        mx = fmaxf(mx, x[src + dh * kHW + dw]);
    }
  xph[t] = (_Float16)mx;
}

// ---------------------------------------------------------------------------
// prep_weights: BFP-quantize W per (cout, cin-block, tap) exactly like bfp()
// (amax over 32 cin per tap); write fp16 tiles [cB][cb][co32][tap][ci32] so a
// conv (cB,cb) W-tile is one contiguous chunk (TDM friendly).
// Also folds BatchNorm into per-channel (scale, bias).
// ---------------------------------------------------------------------------
__global__ void prep_weights_kernel(const float* __restrict__ W,
                                    const float* __restrict__ b,
                                    const float* __restrict__ g,
                                    const float* __restrict__ be,
                                    const float* __restrict__ m,
                                    const float* __restrict__ v,
                                    _Float16* __restrict__ Wh,
                                    float* __restrict__ sb,
                                    int Cout, int Cin, int KS) {
  int t = blockIdx.x * blockDim.x + threadIdx.x;
  if (t < Cout) {
    float inv = rsqrtf(v[t] + 1e-5f);
    float s = g[t] * inv;
    sb[t] = s;
    sb[Cout + t] = be[t] + s * (b[t] - m[t]);
  }
  int KS2 = KS * KS;
  int nCB = Cin >> 5;
  int total = Cout * nCB * KS2;
  if (t >= total) return;
  int tap = t % KS2;
  int cb  = (t / KS2) % nCB;
  int co  = t / (KS2 * nCB);
  float wv[32];
  float amax = 0.f;
#pragma unroll 4
  for (int j = 0; j < 32; ++j) {
    float xv = W[(size_t)(co * Cin + cb * 32 + j) * KS2 + tap];
    wv[j] = xv;
    amax = fmaxf(amax, fabsf(xv));
  }
  float e  = floorf(log2f(fmaxf(amax, 1e-30f)));
  float sc = exp2f(e) * (1.0f / 127.0f);
  float rs = 127.0f * exp2f(-e);
#pragma unroll 4
  for (int j = 0; j < 32; ++j) {
    float q = fminf(fmaxf(roundf(wv[j] * rs), -127.f), 127.f) * sc;
    size_t dst = ((((size_t)(co >> 5) * nCB + cb) * 32 + (co & 31)) * KS2 + tap) * 32 + j;
    Wh[dst] = (_Float16)q;
  }
}

// ---------------------------------------------------------------------------
// Fused conv(KSxKS implicit GEMM) + BN + ReLU + BFP activation quantization.
// Workgroup (8 wave32): M=32 couts (one BFP block), N=128 positions
// (4 output rows x 32 padded cols). Wave w: wm=w&1 (M half), wq=w>>1 owns
// output row wq with TWO 16x16 accumulators sharing the A fragment.
// Double-buffered LDS; TDM DMAs stage the next K-block while WMMAs run:
//   waves 0..R-1 each issue one row TDM, wave 7 issues the W-tile TDM;
//   s_wait_tensorcnt(1) (in-order per wave) guarantees the current tile landed.
// Halo cols / OOB rows are zeroed once in the prologue (constant across K).
// ---------------------------------------------------------------------------
template <int KS>
__global__ void __launch_bounds__(256)
conv_bn_relu_bfp_kernel(const _Float16* __restrict__ Xt,  // [n][cbIn][784][32] fp16
                        const _Float16* __restrict__ Wh,  // [cB][cb][32][KS2][32] fp16
                        const float* __restrict__ sb,     // [scale[Cout] | bias[Cout]]
                        _Float16* __restrict__ Yt,        // repacked fp16 out (or null)
                        float* __restrict__ Yf,           // fp32 NCHW concat out (or null)
                        int Cin, int Cout, int coutOff) {
  constexpr int KS2 = KS * KS;
  constexpr int PAD = (KS - 1) / 2;
  constexpr int R   = 4 + KS - 1;       // staged input rows (4 out rows + halo)
  constexpr int CC  = 32 + 2 * PAD;     // staged input cols
  constexpr int WTILE8 = KS2 * 256;     // W tile in 8-byte units (32*KS2*32*2/8)
  constexpr int ROW8   = kHW * 32 * 2 / 8;  // one interior row = 224 8-byte units

  __shared__ _Float16 sX[2][R][CC][32];   // [buf][row][col][ci]
  __shared__ _Float16 sW[2][32][KS2][32]; // [buf][co][tap][ci]
  __shared__ float sAmax[2][128];
  __shared__ float sScale[32];
  __shared__ float sBias[32];

  const int tid  = threadIdx.x;
  const int lane = tid & 31;
  const int wave = tid >> 5;
  const int wm   = wave & 1;            // M half
  const int wq   = wave >> 1;           // output row within tile (0..3)

  const int h0 = blockIdx.x;            // output rows 4*h0 .. 4*h0+3
  const int cB = blockIdx.y;            // cout block of 32
  const int n  = blockIdx.z;

  if (tid < 32) {
    sScale[tid] = sb[cB * 32 + tid];
    sBias[tid]  = sb[Cout + cB * 32 + tid];
  }

  const int ln    = lane & 15;
  const int mhalf = (lane >> 4) * 8;    // D rows: acc[r] -> cout wm*16+mhalf+r
  const int k0b   = (lane & 16) ? 16 : 0;

  v8f acc[2] = {{}, {}};

  const int nCB = Cin >> 5;
  const size_t xBase = ((size_t)n * nCB) * kPos * 32;        // halves
  const size_t wBase = (size_t)cB * nCB * (WTILE8 * 4);      // halves

  // --- prologue: zero both LDS X buffers (halo stays zero for all K steps) ---
  {
    v8h z = {};
    _Float16* xs = &sX[0][0][0][0];
    for (int ch = tid; ch < 2 * R * CC * 4; ch += 256)
      *((v8h*)xs + ch) = z;
  }
  __syncthreads();  // zeros strictly precede the first DMA writes

  // --- staging: TDM issue (or cooperative vector copy fallback) -------------
  auto issue_stage = [&](int cb, int bu) {
#if USE_TDM
    if (wave < R) {                     // wave r stages input row r
      int hi = h0 * 4 - PAD + wave;
      if ((unsigned)hi < (unsigned)kHW)
        tdm_load_1d(Xt + xBase + (size_t)cb * kPos * 32 + (size_t)hi * kHW * 32,
                    &sX[bu][wave][PAD][0], ROW8);
    } else if (wave == 7) {             // wave 7 stages the W tile
      tdm_load_1d(Wh + wBase + (size_t)cb * (WTILE8 * 4), &sW[bu][0][0][0], WTILE8);
    }
#else
    const _Float16* xsrc = Xt + xBase + (size_t)cb * kPos * 32;
    for (int ch = tid; ch < R * (ROW8 * 2); ch += 256) {
      int rr = ch / (ROW8 * 2);
      int k  = ch % (ROW8 * 2);
      int hi = h0 * 4 - PAD + rr;
      if ((unsigned)hi < (unsigned)kHW)
        *((v8h*)&sX[bu][rr][PAD][0] + k) =
            *((const v8h*)(xsrc + (size_t)hi * kHW * 32) + k);
    }
    const _Float16* wsrc = Wh + wBase + (size_t)cb * (WTILE8 * 4);
    for (int ch = tid; ch < WTILE8 * 2; ch += 256)
      *((v8h*)&sW[bu][0][0][0] + ch) = *((const v8h*)wsrc + ch);
#endif
  };

#if USE_TDM
  const bool issuer =
      (wave < R && (unsigned)(h0 * 4 - PAD + wave) < (unsigned)kHW) || (wave == 7);
#endif

  issue_stage(0, 0);

  for (int cb = 0; cb < nCB; ++cb) {
    const int bu = cb & 1;
    if (cb + 1 < nCB) issue_stage(cb + 1, bu ^ 1);
#if USE_TDM
    if (issuer) {                        // TDMs retire in order per wave
      if (cb + 1 < nCB) __builtin_amdgcn_s_wait_tensorcnt(1);
      else              __builtin_amdgcn_s_wait_tensorcnt(0);
    }
#endif
    __syncthreads();                     // tile cb ready for everyone

    const _Float16* xb = &sX[bu][0][0][0];
    const _Float16* wb = &sW[bu][0][0][0];
#pragma unroll
    for (int tap = 0; tap < KS2; ++tap) {
      const int kh = tap / KS, kw = tap % KS;
      v16h a  = frag_a(wb + (wm * 16 * KS2 + tap) * 32, KS2 * 32, lane);
      v16h b0 = *(const v16h*)(xb + ((wq + kh) * CC + (ln + kw)) * 32 + k0b);
      v16h b1 = *(const v16h*)(xb + ((wq + kh) * CC + (ln + 16 + kw)) * 32 + k0b);
      acc[0] = __builtin_amdgcn_wmma_f32_16x16x32_f16(false, a, false, b0,
                                                      (short)0, acc[0], false, false);
      acc[1] = __builtin_amdgcn_wmma_f32_16x16x32_f16(false, a, false, b1,
                                                      (short)0, acc[1], false, false);
    }
    __syncthreads();                     // all reads of buf done before reuse
  }

  // --- epilogue: BN + ReLU, BFP (shared exponent per 32-ch block per position)
  float yv[2][8];
  float av[2] = {0.f, 0.f};
#pragma unroll
  for (int j = 0; j < 2; ++j) {
#pragma unroll
    for (int r = 0; r < 8; ++r) {
      int cl = wm * 16 + mhalf + r;
      float y = sScale[cl] * acc[j][r] + sBias[cl];
      y = fmaxf(y, 0.f);
      yv[j][r] = y;
      av[j] = fmaxf(av[j], y);          // |y| == y after ReLU
    }
    av[j] = fmaxf(av[j], __shfl_xor(av[j], 16, 32)); // combine M halves (same col)
  }
  const int p0 = wq * 32 + ln;
  if (lane < 16) { sAmax[wm][p0] = av[0]; sAmax[wm][p0 + 16] = av[1]; }
  __syncthreads();

  const int h = h0 * 4 + wq;
  const int nCBo = Cout >> 5;
#pragma unroll
  for (int j = 0; j < 2; ++j) {
    int ocol = ln + j * 16;
    float am = fmaxf(sAmax[0][p0 + j * 16], sAmax[1][p0 + j * 16]);
    float e  = floorf(log2f(fmaxf(am, 1e-30f)));
    float sc = exp2f(e) * (1.0f / 127.0f);
    float rs = 127.0f * exp2f(-e);
    if (ocol < kHW) {
      size_t pos = (size_t)h * kHW + ocol;
      if (Yt) {  // repacked fp16: 8 consecutive channels -> one b128 store
        v8h o;
#pragma unroll
        for (int r = 0; r < 8; ++r)
          o[r] = (_Float16)(fminf(roundf(yv[j][r] * rs), 127.0f) * sc);
        *(v8h*)&Yt[(((size_t)n * nCBo + cB) * kPos + pos) * 32 + wm * 16 + mhalf] = o;
      }
      if (Yf) {  // fp32 NCHW concat slice
#pragma unroll
        for (int r = 0; r < 8; ++r) {
          int cg = cB * 32 + wm * 16 + mhalf + r;
          float q = fminf(roundf(yv[j][r] * rs), 127.0f) * sc;
          Yf[((size_t)n * kCoutTotal + coutOff + cg) * kPos + pos] = q;
        }
      }
    }
  }
}

// ---------------------------------------------------------------------------
// Host launcher
// ---------------------------------------------------------------------------
extern "C" void kernel_launch(void* const* d_in, const int* in_sizes, int n_in,
                              void* d_out, int out_size, void* d_ws, size_t ws_size,
                              hipStream_t stream) {
  (void)in_sizes; (void)n_in; (void)out_size; (void)ws_size;

  struct LayerCfg { int cout, cin, ks; };
  const LayerCfg L[7] = {
      {256, 512, 1},  // b1c1
      {160, 512, 1},  // b2c1
      {320, 160, 3},  // b2c2
      {32,  512, 1},  // b3c1
      {128, 32,  3},  // b3c2
      {128, 128, 3},  // b3c3
      {128, 512, 1},  // b4c1
  };

  const float* x = (const float*)d_in[0];

  char* ws = (char*)d_ws;
  size_t off = 0;
  auto alloc = [&](size_t bytes) {
    size_t o = off;
    off = (off + bytes + 255) & ~(size_t)255;
    return (char*)(ws + o);
  };

  _Float16* xh  = (_Float16*)alloc((size_t)kImgs * 512 * kPos * sizeof(_Float16));
  _Float16* xph = (_Float16*)alloc((size_t)kImgs * 512 * kPos * sizeof(_Float16));
  _Float16* y2a = (_Float16*)alloc((size_t)kImgs * 160 * kPos * sizeof(_Float16));
  _Float16* y3a = (_Float16*)alloc((size_t)kImgs * 32  * kPos * sizeof(_Float16));
  _Float16* y3b = (_Float16*)alloc((size_t)kImgs * 128 * kPos * sizeof(_Float16));

  _Float16* Whp[7];
  float*    sbp[7];
  for (int i = 0; i < 7; ++i) {
    size_t wElems = (size_t)L[i].cout * L[i].ks * L[i].ks * L[i].cin;
    Whp[i] = (_Float16*)alloc(wElems * sizeof(_Float16));
    sbp[i] = (float*)alloc((size_t)2 * L[i].cout * sizeof(float));
  }

  // 1) input conversion (repacked) + maxpool branch input
  {
    int total = kImgs * 512 * kPos;
    prep_input_kernel<<<(total + 255) / 256, 256, 0, stream>>>(x, xh, xph, total);
  }

  // 2) weight BFP quantization + BN folding
  for (int i = 0; i < 7; ++i) {
    const float* W  = (const float*)d_in[1 + i * 6 + 0];
    const float* b  = (const float*)d_in[1 + i * 6 + 1];
    const float* g  = (const float*)d_in[1 + i * 6 + 2];
    const float* be = (const float*)d_in[1 + i * 6 + 3];
    const float* m  = (const float*)d_in[1 + i * 6 + 4];
    const float* v  = (const float*)d_in[1 + i * 6 + 5];
    int ks2 = L[i].ks * L[i].ks;
    int total = L[i].cout * (L[i].cin / 32) * ks2;
    if (total < L[i].cout) total = L[i].cout;
    prep_weights_kernel<<<(total + 255) / 256, 256, 0, stream>>>(
        W, b, g, be, m, v, Whp[i], sbp[i], L[i].cout, L[i].cin, L[i].ks);
  }

  float* out = (float*)d_out;
  auto conv = [&](int li, const _Float16* Xin, _Float16* Yt, float* Yf, int coutOff) {
    dim3 grid(kHW / 4, L[li].cout / 32, kImgs);
    if (L[li].ks == 1) {
      conv_bn_relu_bfp_kernel<1><<<grid, 256, 0, stream>>>(
          Xin, Whp[li], sbp[li], Yt, Yf, L[li].cin, L[li].cout, coutOff);
    } else {
      conv_bn_relu_bfp_kernel<3><<<grid, 256, 0, stream>>>(
          Xin, Whp[li], sbp[li], Yt, Yf, L[li].cin, L[li].cout, coutOff);
    }
  };

  // 3) branches (stream-ordered dependencies)
  conv(0, xh,  nullptr, out, 0);    // b1c1 -> concat[0:256]
  conv(1, xh,  y2a, nullptr, 0);    // b2c1 -> y2a
  conv(2, y2a, nullptr, out, 256);  // b2c2 -> concat[256:576]
  conv(3, xh,  y3a, nullptr, 0);    // b3c1 -> y3a
  conv(4, y3a, y3b, nullptr, 0);    // b3c2 -> y3b
  conv(5, y3b, nullptr, out, 576);  // b3c3 -> concat[576:704]
  conv(6, xph, nullptr, out, 704);  // b4c1(maxpool(x)) -> concat[704:832]
}
